// HeteroRGCNLayer_28209345200161
// MI455X (gfx1250) — matrix operand order repaired and checked by reading, not compiled, not run on hardware
//
#include <hip/hip_runtime.h>

// ---------------------------------------------------------------------------
// HeteroRGCN layer for MI455X (gfx1250, wave32, WMMA)
//   per relation r: Wh = x @ Wr^T + br ; out += scatter_mean(Wh[src], dst)
// v3: bf16 pre-convert + each wave computes a 32x64 strip (2 M-tiles x
//     4 N-tiles = 8 independent v_wmma accumulators). Each B fragment feeds
//     two WMMAs (halves B load traffic per FLOP) and the deep independent
//     WMMA chain lets the scheduler overlap loads instead of full waits.
// ---------------------------------------------------------------------------

typedef __attribute__((ext_vector_type(16))) __bf16 v16bf;
typedef __attribute__((ext_vector_type(8)))  __bf16 v8bf;
typedef __attribute__((ext_vector_type(8)))  float  v8f;

#define IN_SIZE  512
#define OUT_SIZE 512

struct bfrag { int4 lo, hi; };   // 32 bytes == v16bf

// A fragment (16x32 bf16, ISA 7.12.2): elements 0..7 -> K=k0..k0+7,
// elements 8..15 -> K=k0+16..k0+23  (k0 already includes lane-group offset)
__device__ __forceinline__ v16bf load_a_frag(const __bf16* row, int k0) {
    bfrag f;
    f.lo = *(const int4*)(row + k0);
    f.hi = *(const int4*)(row + k0 + 16);
    return __builtin_bit_cast(v16bf, f);
}

// B fragment (32x16 bf16): elements 0..7 -> K=k0..k0+7, elements 8..15 -> K=k0+16..k0+23
__device__ __forceinline__ v16bf load_b_frag(const __bf16* row, int k0) {
    bfrag f;
    f.lo = *(const int4*)(row + k0);
    f.hi = *(const int4*)(row + k0 + 16);
    return __builtin_bit_cast(v16bf, f);
}

__device__ __forceinline__ v8f wmma_bf16(v16bf a, v16bf b, v8f c) {
    return __builtin_amdgcn_wmma_f32_16x16x32_bf16(
        false, a, false, b, (short)0, c, false, false);
}

// f32 -> bf16 bulk convert, 8 elements per thread
__global__ __launch_bounds__(256) void cvt_f32_to_bf16(
    const float* __restrict__ in, __bf16* __restrict__ out, long long n)
{
    const long long i = ((long long)blockIdx.x * blockDim.x + threadIdx.x) * 8;
    if (i >= n) return;
    const float4 p0 = *(const float4*)(in + i);
    const float4 p1 = *(const float4*)(in + i + 4);
    v8bf o;
    o[0] = (__bf16)p0.x; o[1] = (__bf16)p0.y; o[2] = (__bf16)p0.z; o[3] = (__bf16)p0.w;
    o[4] = (__bf16)p1.x; o[5] = (__bf16)p1.y; o[6] = (__bf16)p1.z; o[7] = (__bf16)p1.w;
    *(v8bf*)(out + i) = o;
}

// Wh = xb @ wb^T + bias.  grid = ceil(n_nodes/32), block = 256 (8 waves).
// Wave w: rows [32*blockIdx.x, +32) x cols [64*w, +64) -> 8 WMMA tiles.
__global__ __launch_bounds__(256) void rgcn_gemm_bf16_wmma(
    const __bf16* __restrict__ xb,   // [n_nodes, IN_SIZE] bf16
    const __bf16* __restrict__ wb,   // [OUT_SIZE, IN_SIZE] bf16 (row = out feat)
    const float*  __restrict__ bias, // [OUT_SIZE]
    float* __restrict__ Wh,          // [n_nodes, OUT_SIZE]
    int n_nodes)
{
    const int lane  = threadIdx.x & 31;
    const int wave  = threadIdx.x >> 5;     // 0..7
    const int l15   = lane & 15;
    const int lg    = lane >> 4;            // lane group 0/1
    const int tm    = blockIdx.x;           // 32-row strip
    const int nbase = wave * 64;

    // Clamp A-row loads (last strip may exceed n_nodes); EXEC stays all-ones.
    const int row0 = tm * 32 +  0 + l15;
    const int row1 = tm * 32 + 16 + l15;
    const int ar0  = row0 < n_nodes ? row0 : n_nodes - 1;
    const int ar1  = row1 < n_nodes ? row1 : n_nodes - 1;

    const __bf16* __restrict__ x0 = xb + (size_t)ar0 * IN_SIZE;
    const __bf16* __restrict__ x1 = xb + (size_t)ar1 * IN_SIZE;
    const __bf16* __restrict__ w0 = wb + (size_t)(nbase +  0 + l15) * IN_SIZE;
    const __bf16* __restrict__ w1 = wb + (size_t)(nbase + 16 + l15) * IN_SIZE;
    const __bf16* __restrict__ w2 = wb + (size_t)(nbase + 32 + l15) * IN_SIZE;
    const __bf16* __restrict__ w3 = wb + (size_t)(nbase + 48 + l15) * IN_SIZE;

    v8f c00 = {}, c01 = {}, c02 = {}, c03 = {};   // M-tile 0 x N-tiles 0..3
    v8f c10 = {}, c11 = {}, c12 = {}, c13 = {};   // M-tile 1 x N-tiles 0..3

    #pragma unroll 2
    for (int kk = 0; kk < IN_SIZE; kk += 32) {
        const int ka = kk + lg * 8;
        const int kb = kk + lg * 8;
        const v16bf a0 = load_a_frag(x0, ka);
        const v16bf a1 = load_a_frag(x1, ka);
        const v16bf b0 = load_b_frag(w0, kb);
        const v16bf b1 = load_b_frag(w1, kb);
        const v16bf b2 = load_b_frag(w2, kb);
        const v16bf b3 = load_b_frag(w3, kb);
        c00 = wmma_bf16(a0, b0, c00);  c10 = wmma_bf16(a1, b0, c10);
        c01 = wmma_bf16(a0, b1, c01);  c11 = wmma_bf16(a1, b1, c11);
        c02 = wmma_bf16(a0, b2, c02);  c12 = wmma_bf16(a1, b2, c12);
        c03 = wmma_bf16(a0, b3, c03);  c13 = wmma_bf16(a1, b3, c13);
    }

    // D layout: element r -> row (mbase + r + lg*8), col (nbase + t*16 + l15)
    const float bv0 = bias[nbase +  0 + l15];
    const float bv1 = bias[nbase + 16 + l15];
    const float bv2 = bias[nbase + 32 + l15];
    const float bv3 = bias[nbase + 48 + l15];
    #pragma unroll
    for (int r = 0; r < 8; ++r) {
        const int orow0 = tm * 32 + r + lg * 8;        // M-tile 0
        const int orow1 = orow0 + 16;                  // M-tile 1
        if (orow0 < n_nodes) {
            float* o = Wh + (size_t)orow0 * OUT_SIZE;
            o[nbase +  0 + l15] = c00[r] + bv0;
            o[nbase + 16 + l15] = c01[r] + bv1;
            o[nbase + 32 + l15] = c02[r] + bv2;
            o[nbase + 48 + l15] = c03[r] + bv3;
        }
        if (orow1 < n_nodes) {
            float* o = Wh + (size_t)orow1 * OUT_SIZE;
            o[nbase +  0 + l15] = c10[r] + bv0;
            o[nbase + 16 + l15] = c11[r] + bv1;
            o[nbase + 32 + l15] = c12[r] + bv2;
            o[nbase + 48 + l15] = c13[r] + bv3;
        }
    }
}

// In-degree accumulation: cnt[dst[e]] += 1
__global__ __launch_bounds__(256) void rgcn_degree(
    const int* __restrict__ dst, float* __restrict__ cnt, int n_edges)
{
    const int e = blockIdx.x * blockDim.x + threadIdx.x;
    if (e < n_edges) atomicAdd(&cnt[dst[e]], 1.0f);
}

// cnt[i] = 1 / max(cnt[i], 1)
__global__ __launch_bounds__(256) void rgcn_invdeg(
    float* __restrict__ cnt, int n_nodes)
{
    const int i = blockIdx.x * blockDim.x + threadIdx.x;
    if (i < n_nodes) cnt[i] = 1.0f / fmaxf(cnt[i], 1.0f);
}

// One block per edge: out[dst] += Wh[src] * invdeg[dst]   (128 lanes x float4)
__global__ __launch_bounds__(128) void rgcn_edge_scatter(
    const float* __restrict__ Wh, const int* __restrict__ src,
    const int* __restrict__ dst, const float* __restrict__ invdeg,
    float* __restrict__ out)
{
    const int e = blockIdx.x;
    const int s = src[e];
    const int d = dst[e];
    const float scale = invdeg[d];
    const int f = threadIdx.x * 4;

    const float4 v = *(const float4*)(Wh + (size_t)s * OUT_SIZE + f);
    float* o = out + (size_t)d * OUT_SIZE + f;
    atomicAdd(o + 0, v.x * scale);
    atomicAdd(o + 1, v.y * scale);
    atomicAdd(o + 2, v.z * scale);
    atomicAdd(o + 3, v.w * scale);
}

extern "C" void kernel_launch(void* const* d_in, const int* in_sizes, int n_in,
                              void* d_out, int out_size, void* d_ws, size_t ws_size,
                              hipStream_t stream) {
    const float* x  = (const float*)d_in[0];
    const float* Wr[3]   = { (const float*)d_in[1], (const float*)d_in[3], (const float*)d_in[5] };
    const float* br[3]   = { (const float*)d_in[2], (const float*)d_in[4], (const float*)d_in[6] };
    const int*   srcr[3] = { (const int*)d_in[7],  (const int*)d_in[9],  (const int*)d_in[11] };
    const int*   dstr[3] = { (const int*)d_in[8],  (const int*)d_in[10], (const int*)d_in[12] };

    const int n_nodes = in_sizes[0] / IN_SIZE;   // 50000
    const int n_edges = in_sizes[7];             // 150000

    float* out = (float*)d_out;

    // ws layout: Wh [n_nodes*512 f32] | cnt [n_nodes f32] | xb [n_nodes*512 bf16]
    //            | wb [3*512*512 bf16]
    float*  Wh  = (float*)d_ws;
    float*  cnt = Wh + (size_t)n_nodes * OUT_SIZE;
    __bf16* xb  = (__bf16*)(cnt + n_nodes);
    __bf16* wb  = xb + (size_t)n_nodes * IN_SIZE;

    // out = 0
    hipMemsetAsync(out, 0, (size_t)n_nodes * OUT_SIZE * sizeof(float), stream);

    // Pre-convert x and all three W matrices to bf16 (one bandwidth pass)
    {
        const long long nx = (long long)n_nodes * IN_SIZE;
        const long long nw = (long long)OUT_SIZE * IN_SIZE;
        cvt_f32_to_bf16<<<(int)((nx / 8 + 255) / 256), 256, 0, stream>>>(x, xb, nx);
        for (int r = 0; r < 3; ++r)
            cvt_f32_to_bf16<<<(int)((nw / 8 + 255) / 256), 256, 0, stream>>>(
                Wr[r], wb + (size_t)r * OUT_SIZE * IN_SIZE, nw);
    }

    const int gemm_blocks = (n_nodes + 31) / 32;   // 32-row strips
    const int deg_blocks  = (n_edges + 255) / 256;
    const int inv_blocks  = (n_nodes + 255) / 256;

    for (int r = 0; r < 3; ++r) {
        rgcn_gemm_bf16_wmma<<<gemm_blocks, 256, 0, stream>>>(
            xb, wb + (size_t)r * OUT_SIZE * IN_SIZE, br[r], Wh, n_nodes);
        hipMemsetAsync(cnt, 0, (size_t)n_nodes * sizeof(float), stream);
        rgcn_degree<<<deg_blocks, 256, 0, stream>>>(dstr[r], cnt, n_edges);
        rgcn_invdeg<<<inv_blocks, 256, 0, stream>>>(cnt, n_nodes);
        rgcn_edge_scatter<<<n_edges, 128, 0, stream>>>(Wh, srcr[r], dstr[r], cnt, out);
    }
}
